// WindowAttention_Sparse_6502580486252
// MI455X (gfx1250) — matrix operand order repaired and checked by reading, not compiled, not run on hardware
//
#include <hip/hip_runtime.h>
#include <hip/hip_bf16.h>

typedef __attribute__((ext_vector_type(16))) _Float16 v16h;
typedef __attribute__((ext_vector_type(8)))  _Float16 v8h;
typedef __attribute__((ext_vector_type(8)))  float    v8f;

#define WMMA(a, b, c) __builtin_amdgcn_wmma_f32_16x16x32_f16(false, (a), false, (b), (short)0, (c), false, false)

// B-operand: 16 contiguous f16 per lane (col = lane&15, K-run picked by lane>>4; caller bakes both into p)
__device__ __forceinline__ v16h ld_b16(const _Float16* p) {
    v8h lo = *(const v8h*)p;
    v8h hi = *(const v8h*)(p + 8);
    return __builtin_shufflevector(lo, hi, 0,1,2,3,4,5,6,7,8,9,10,11,12,13,14,15);
}
// A-operand: 16 contiguous f16 per lane (K = k0 + e + 16*(lane>>4))
__device__ __forceinline__ v16h ld_a16(const _Float16* p) {
    v8h lo = *(const v8h*)p;
    v8h hi = *(const v8h*)(p + 8);
    return __builtin_shufflevector(lo, hi, 0,1,2,3,4,5,6,7,8,9,10,11,12,13,14,15);
}

// ---- prep: transpose + f16-convert weights, fuse qkv bias, expand per-head relpos bias ----
__global__ void prep_weights(const float* __restrict__ wq, const float* __restrict__ wkv,
                             const float* __restrict__ bq, const float* __restrict__ bkv,
                             const float* __restrict__ projw,
                             _Float16* __restrict__ wt_qkv, _Float16* __restrict__ proj_wt,
                             float* __restrict__ bqkv) {
    int i = blockIdx.x * 256 + threadIdx.x;
    if (i < 768 * 256) {                      // wt_qkv[n][c] = W[c][n]
        int n = i >> 8, c = i & 255;
        float v = (n < 256) ? wq[c * 256 + n] : wkv[c * 512 + (n - 256)];
        wt_qkv[i] = (_Float16)v;
    }
    if (i < 256 * 256) {                      // proj_wt[n][c] = proj_w[c][n]
        int n = i >> 8, c = i & 255;
        proj_wt[i] = (_Float16)projw[c * 256 + n];
    }
    if (i < 768) bqkv[i] = (i < 256) ? bq[i] : bkv[i - 256];
}

__global__ void prep_bias(const float* __restrict__ table, const int* __restrict__ rel,
                          float* __restrict__ bexp) {
    int i = blockIdx.x * 256 + threadIdx.x;   // 8*64*64
    if (i < 8 * 64 * 64) {
        int h = i >> 12, qk = i & 4095;
        bexp[i] = table[rel[qk] * 8 + h];
    }
}

// ---- main: one workgroup per window, 8 waves, wave-per-head attention ----
__global__ __launch_bounds__(256)
void win_attn_kernel(const float* __restrict__ x,
                     const _Float16* __restrict__ wt_qkv,
                     const float* __restrict__ bqkv,
                     const _Float16* __restrict__ proj_wt,
                     const float* __restrict__ projb,
                     const float* __restrict__ bexp_g,
                     const float* __restrict__ gw,
                     float* __restrict__ out) {
    extern __shared__ _Float16 smem[];
    _Float16* QK = smem;                       // [64][512] Q|K, Q cols reused for O
    _Float16* Vt = smem + 64 * 512;            // [256][64] V transposed (dh-major)
    _Float16* Xs = smem + 64 * 512 + 256 * 64; // [64][256] stage A only
    _Float16* At = Xs;                         // [8][64][64] attn, stage B (reuses Xs)

    const int tid  = threadIdx.x;
    const int lane = tid & 31;
    const int wv   = tid >> 5;       // wave id 0..7 (= head in stage B)
    const int lo16 = lane & 15;
    const int hiH  = lane >> 4;      // 0/1 lane-half
    const int blk  = blockIdx.x;
    const float scale = 0.17677669529663687f;  // 1/sqrt(32)

    // X -> LDS f16
    const float* xp = x + (size_t)blk * (64 * 256);
    for (int i = tid; i < 64 * 256; i += 256) Xs[i] = (_Float16)xp[i];
    __syncthreads();

    // ---- Stage A: QKV = X @ W^T (+bias, q*scale); wave owns 96 cols ----
    for (int nt = 0; nt < 6; ++nt) {
        const int col = wv * 96 + nt * 16 + lo16;
        v8f acc[4] = {};
        #pragma unroll
        for (int c0 = 0; c0 < 256; c0 += 32) {
            v16h b = ld_b16(wt_qkv + col * 256 + c0 + (hiH << 4));
            const _Float16* ap = Xs + lo16 * 256 + c0 + (hiH << 4);
            #pragma unroll
            for (int mt = 0; mt < 4; ++mt)
                acc[mt] = WMMA(ld_a16(ap + mt * 16 * 256), b, acc[mt]);
        }
        const float bias = bqkv[col];
        const float sc = (col < 256) ? scale : 1.0f;
        #pragma unroll
        for (int mt = 0; mt < 4; ++mt) {
            #pragma unroll
            for (int r = 0; r < 8; ++r) {
                int row = mt * 16 + r + (hiH << 3);
                float v = (acc[mt][r] + bias) * sc;
                if (col < 512) QK[row * 512 + col] = (_Float16)v;
                else           Vt[(col - 512) * 64 + row] = (_Float16)v;  // transposed V
            }
        }
    }
    __syncthreads();

    // mixing weights: softmax over 2
    const float w0 = gw[0], w1 = gw[1];
    const float mw = fmaxf(w0, w1);
    const float e0 = __expf(w0 - mw), e1 = __expf(w1 - mw);
    const float ww0 = e0 / (e0 + e1), ww1 = e1 / (e0 + e1);

    // ---- Stage B: per-head S = Q K^T, bias, softmax/relu^2 mix, attn -> LDS ----
    const int h = wv;
    const float* bexp = bexp_g + h * 4096;
    _Float16* Ah = At + h * 4096;
    for (int mi = 0; mi < 4; ++mi) {
        v16h aq = ld_a16(QK + (mi * 16 + lo16) * 512 + 32 * h + (hiH << 4));
        v8f s[4];
        #pragma unroll
        for (int ni = 0; ni < 4; ++ni) {
            v16h bk = ld_b16(QK + (ni * 16 + lo16) * 512 + 256 + 32 * h + (hiH << 4));
            v8f z = {};
            s[ni] = WMMA(aq, bk, z);
        }
        #pragma unroll
        for (int ni = 0; ni < 4; ++ni) {
            #pragma unroll
            for (int r = 0; r < 8; ++r) {
                int q = mi * 16 + r + (hiH << 3);
                s[ni][r] += bexp[q * 64 + ni * 16 + lo16];
            }
        }
        #pragma unroll
        for (int r = 0; r < 8; ++r) {   // row lives in one 16-lane half across s[0..3]
            float m = fmaxf(fmaxf(s[0][r], s[1][r]), fmaxf(s[2][r], s[3][r]));
            m = fmaxf(m, __shfl_xor(m, 1));
            m = fmaxf(m, __shfl_xor(m, 2));
            m = fmaxf(m, __shfl_xor(m, 4));
            m = fmaxf(m, __shfl_xor(m, 8));
            float ex[4], sum = 0.f;
            #pragma unroll
            for (int ni = 0; ni < 4; ++ni) { ex[ni] = __expf(s[ni][r] - m); sum += ex[ni]; }
            sum += __shfl_xor(sum, 1);
            sum += __shfl_xor(sum, 2);
            sum += __shfl_xor(sum, 4);
            sum += __shfl_xor(sum, 8);
            const float inv = 1.f / sum;
            const int q = mi * 16 + r + (hiH << 3);
            #pragma unroll
            for (int ni = 0; ni < 4; ++ni) {
                float sv = s[ni][r];
                float rl = fmaxf(sv, 0.f);
                float a = ww0 * (ex[ni] * inv) + ww1 * rl * rl;
                Ah[q * 64 + ni * 16 + lo16] = (_Float16)a;
            }
        }
    }
    __syncthreads();

    // ---- O = attn @ V -> dead Q columns of QK ----
    for (int mi = 0; mi < 4; ++mi) {
        v8f o[2] = {};
        #pragma unroll
        for (int kc = 0; kc < 2; ++kc) {
            v16h aa = ld_a16(Ah + (mi * 16 + lo16) * 64 + kc * 32 + (hiH << 4));
            #pragma unroll
            for (int nj = 0; nj < 2; ++nj) {
                v16h bv = ld_b16(Vt + (h * 32 + nj * 16 + lo16) * 64 + kc * 32 + (hiH << 4));
                o[nj] = WMMA(aa, bv, o[nj]);
            }
        }
        #pragma unroll
        for (int nj = 0; nj < 2; ++nj) {
            #pragma unroll
            for (int r = 0; r < 8; ++r) {
                int row = mi * 16 + r + (hiH << 3);
                QK[row * 512 + 32 * h + nj * 16 + lo16] = (_Float16)o[nj][r];
            }
        }
    }
    __syncthreads();

    // ---- Stage C: out = O @ proj_w^T + proj_b ----
    float* op = out + (size_t)blk * (64 * 256);
    for (int nt = 0; nt < 2; ++nt) {
        const int col = wv * 32 + nt * 16 + lo16;
        v8f acc[4] = {};
        #pragma unroll
        for (int c0 = 0; c0 < 256; c0 += 32) {
            v16h b = ld_b16(proj_wt + col * 256 + c0 + (hiH << 4));
            #pragma unroll
            for (int mt = 0; mt < 4; ++mt) {
                v16h a = ld_a16(QK + (mt * 16 + lo16) * 512 + c0 + (hiH << 4));
                acc[mt] = WMMA(a, b, acc[mt]);
            }
        }
        const float pb = projb[col];
        #pragma unroll
        for (int mt = 0; mt < 4; ++mt) {
            #pragma unroll
            for (int r = 0; r < 8; ++r) {
                int row = mt * 16 + r + (hiH << 3);
                op[row * 256 + col] = acc[mt][r] + pb;
            }
        }
    }
}

extern "C" void kernel_launch(void* const* d_in, const int* in_sizes, int n_in,
                              void* d_out, int out_size, void* d_ws, size_t ws_size,
                              hipStream_t stream) {
    const float* x      = (const float*)d_in[0];
    const float* wq     = (const float*)d_in[1];
    const float* bq     = (const float*)d_in[2];
    const float* wkv    = (const float*)d_in[3];
    const float* bkv    = (const float*)d_in[4];
    const float* btab   = (const float*)d_in[5];
    const float* gw     = (const float*)d_in[6];
    const float* projw  = (const float*)d_in[7];
    const float* projb  = (const float*)d_in[8];
    const int*   relidx = (const int*)d_in[9];

    char* ws = (char*)d_ws;
    _Float16* wt_qkv  = (_Float16*)(ws);             // 768*256*2  = 393216 B
    _Float16* proj_wt = (_Float16*)(ws + 393216);    // 256*256*2  = 131072 B
    float*    bqkv    = (float*)(ws + 524288);       // 768*4      =   3072 B
    float*    bexp    = (float*)(ws + 527360);       // 8*64*64*4  = 131072 B

    prep_weights<<<768, 256, 0, stream>>>(wq, wkv, bq, bkv, projw, wt_qkv, proj_wt, bqkv);
    prep_bias<<<128, 256, 0, stream>>>(btab, relidx, bexp);

    const size_t smem = (size_t)(64 * 512 + 256 * 64 + 8 * 64 * 64) * sizeof(_Float16); // 160 KB
    (void)hipFuncSetAttribute((const void*)win_attn_kernel,
                              hipFuncAttributeMaxDynamicSharedMemorySize, (int)smem);
    win_attn_kernel<<<4096, 256, smem, stream>>>(x, wt_qkv, bqkv, proj_wt, projb, bexp, gw,
                                                 (float*)d_out);
}
